// TCN_LSTM_7773890805994
// MI455X (gfx1250) — compile-verified, matrix operand order checked by reading, not run on hardware
//
#include <hip/hip_runtime.h>
#include <math.h>

// ---------------------------------------------------------------------------
// TCN+LSTM forecaster for MI455X (gfx1250, wave32, WMMA).
//
// Restructurings:
//  * TCN layers are linear -> collapsed to one (128 x 50) effective filter.
//  * mem-scan LSTM input is constant x0 -> x0@Wih0^T + b0 computed once and
//    kept in REGISTERS (each lane owns exactly the 8 C-elements it re-adds).
//  * Batch rows independent -> 4 persistent workgroups of 16 rows.
//  * All B-matrix (weight) WMMA fragments are loop-invariant per wave ->
//    loaded once into VGPRs before the scan; the 1999-step hot loop touches
//    only LDS (h-state) + WMMA + gate VALU. 4 barriers/step.
// ---------------------------------------------------------------------------

typedef __attribute__((ext_vector_type(16))) _Float16 v16h;
typedef __attribute__((ext_vector_type(8)))  float    v8f;
typedef __attribute__((ext_vector_type(4)))  unsigned u32x4;

#define WMMA_F16(a, b, c) \
  __builtin_amdgcn_wmma_f32_16x16x32_f16(false, (a), false, (b), (short)0, (c), false, false)

// Load a 16x32 (rows x K) 16-bit A/B fragment from a row-major (rows, ldk)
// matrix. ISA layout (CDNA5 ISA §7.12.2): lane l holds row (l&15);
// lanes 0-15 take K = kbase+[0..7] and kbase+[16..23], lanes 16-31 take
// K = kbase+[8..15] and kbase+[24..31]  -> two contiguous 16B loads per lane.
__device__ inline v16h load_frag16(const _Float16* base, int row, int ldk,
                                   int kbase, int half) {
  union { v16h v; u32x4 q[2]; } f;
  const _Float16* p = base + (size_t)row * ldk + kbase + 8 * half;
  f.q[0] = *(const u32x4*)(p);
  f.q[1] = *(const u32x4*)(p + 16);
  return f.v;
}

__device__ inline float sigmoidf_(float x) { return 1.0f / (1.0f + __expf(-x)); }

// ---------------------------------------------------------------------------
// K1: collapse w3 (dil 4) o w2 (dil 2) o w1 (dil 1) into Weff (128 x 50) and
//     effective bias beff (128). One block per output channel.
// ---------------------------------------------------------------------------
__global__ __launch_bounds__(256) void compose_weights(
    const float* __restrict__ w1, const float* __restrict__ cb1,
    const float* __restrict__ w2, const float* __restrict__ cb2,
    const float* __restrict__ w3, const float* __restrict__ cb3,
    float* __restrict__ Weff32, _Float16* __restrict__ Weffh,
    float* __restrict__ beff) {
  const int co  = blockIdx.x;   // 0..127
  const int tid = threadIdx.x;  // 256 threads

  __shared__ float w3row[128 * 8];   // w3[co][c2][k3]
  __shared__ float W32u[128][22];    // sum over {k3,k2 : 2k3+k2=u} of w3*w2

  for (int i = tid; i < 128 * 8; i += 256) w3row[i] = w3[co * 1024 + i];
  __syncthreads();

  // W32u[c1][u] = sum_{k3,k2:2k3+k2=u} sum_{c2} w3[co][c2][k3] * w2[c2][c1][k2]
  for (int idx = tid; idx < 128 * 22; idx += 256) {
    const int c1 = idx / 22, u = idx % 22;
    float s = 0.f;
    for (int k3 = 0; k3 < 8; ++k3) {
      const int k2 = u - 2 * k3;
      if ((unsigned)k2 > 7u) continue;
      float p = 0.f;
      for (int c2 = 0; c2 < 128; ++c2)
        p += w3row[c2 * 8 + k3] * w2[(c2 * 128 + c1) * 8 + k2];
      s += p;
    }
    W32u[c1][u] = s;
  }
  __syncthreads();

  // Weff[co][tau] = sum_{u,k1:2u+k1=tau} sum_{c1} W32u[c1][u] * w1[c1][k1]
  if (tid < 64) {
    const int tau = tid;
    float s = 0.f;
    if (tau < 50) {
      for (int u = 0; u < 22; ++u) {
        const int k1 = tau - 2 * u;
        if ((unsigned)k1 > 7u) continue;
        for (int c1 = 0; c1 < 128; ++c1) s += W32u[c1][u] * w1[c1 * 8 + k1];
      }
      Weff32[co * 50 + tau] = s;
    }
    Weffh[co * 64 + tau] = (tau < 50) ? (_Float16)s : (_Float16)0.f;  // K padded to 64
  }

  // beff[co] = cb3 + sum w3*cb2 + sum (sum_u W32u)*cb1
  if (tid == 0) {
    float s = cb3[co];
    for (int c2 = 0; c2 < 128; ++c2) {
      float ws = 0.f;
      for (int k3 = 0; k3 < 8; ++k3) ws += w3row[c2 * 8 + k3];
      s += ws * cb2[c2];
    }
    for (int c1 = 0; c1 < 128; ++c1) {
      float ws = 0.f;
      for (int u = 0; u < 22; ++u) ws += W32u[c1][u];
      s += ws * cb1[c1];
    }
    beff[co] = s;
  }
}

// ---------------------------------------------------------------------------
// K2: f32 -> f16 weight conversion (row-major (512,128) is already (N,K)) and
//     LSTM bias fusion b = bih + bhh.
// ---------------------------------------------------------------------------
__global__ __launch_bounds__(256) void convert_lstm(
    const float* __restrict__ Wih0, const float* __restrict__ Whh0,
    const float* __restrict__ Wih1, const float* __restrict__ Whh1,
    const float* __restrict__ bih0, const float* __restrict__ bhh0,
    const float* __restrict__ bih1, const float* __restrict__ bhh1,
    _Float16* __restrict__ o0, _Float16* __restrict__ o1,
    _Float16* __restrict__ o2, _Float16* __restrict__ o3,
    float* __restrict__ b0, float* __restrict__ b1) {
  const int i = blockIdx.x * blockDim.x + threadIdx.x;
  if (i < 512 * 128) {
    o0[i] = (_Float16)Wih0[i];
    o1[i] = (_Float16)Whh0[i];
    o2[i] = (_Float16)Wih1[i];
    o3[i] = (_Float16)Whh1[i];
  }
  if (i < 512) {
    b0[i] = bih0[i] + bhh0[i];
    b1[i] = bih1[i] + bhh1[i];
  }
}

// ---------------------------------------------------------------------------
// K3: persistent memory scan. grid=4 blocks x 1024 threads (32 waves).
// Block g owns batch rows [16g, 16g+16). Wave w owns gate columns
// [16w, 16w+16) of N=512. Weight fragments register-resident: the steady
// state per step is 12 WMMAs fed purely from LDS h-state. 4 barriers/step.
// Emits the last FOV=50 scalar predictions per row -> winbuf.
// ---------------------------------------------------------------------------
__global__ __launch_bounds__(1024) void phase_b(
    const float* __restrict__ X, int L, int T,
    const float* __restrict__ Weff32, const float* __restrict__ beff,
    const _Float16* __restrict__ Wih0h, const _Float16* __restrict__ Whh0h,
    const _Float16* __restrict__ Wih1h, const _Float16* __restrict__ Whh1h,
    const float* __restrict__ b0, const float* __restrict__ b1,
    const float* __restrict__ Wl, const float* __restrict__ bl,
    float* __restrict__ winbuf) {
  __shared__ __align__(16) float    gates[16][512];
  __shared__ __align__(16) _Float16 h0buf[16][128];
  __shared__ __align__(16) _Float16 h1buf[16][128];
  __shared__ __align__(16) _Float16 afeat[16][128];  // x0 as f16 A-matrix
  __shared__ float c0s[16][128], c1s[16][128];
  __shared__ float wls[128];

  const int tid   = threadIdx.x;
  const int lane  = tid & 31, l15 = lane & 15, half = lane >> 4;
  const int wave  = tid >> 5;              // 0..31 == N-tile
  const int b0row = blockIdx.x * 16;       // global batch base
  const int n     = wave * 16 + l15;       // this lane's gate column

  // x0 = relu(Weff @ X[:, 0:50] + beff)   (only the first 50 samples matter)
  for (int e = tid; e < 16 * 128; e += 1024) {
    const int bb = e >> 7, co = e & 127;
    const float* xr = X + (size_t)(b0row + bb) * L;
    float s = beff[co];
#pragma unroll 10
    for (int t = 0; t < 50; ++t) s += Weff32[co * 50 + t] * xr[t];
    afeat[bb][co] = (_Float16)(s > 0.f ? s : 0.f);
  }
  for (int e = tid; e < 16 * 128; e += 1024) {
    const int bb = e >> 7, hh = e & 127;
    h0buf[bb][hh] = (_Float16)0.f;
    h1buf[bb][hh] = (_Float16)0.f;
    c0s[bb][hh] = 0.f;
    c1s[bb][hh] = 0.f;
  }
  if (tid < 128) wls[tid] = Wl[tid];
  __syncthreads();

  // Loop-invariant weight fragments -> VGPRs (96 VGPRs/lane).
  v16h wf[12];
#pragma unroll
  for (int kb = 0; kb < 4; ++kb) {
    wf[kb]     = load_frag16(Whh0h, n, 128, kb * 32, half);
    wf[4 + kb] = load_frag16(Wih1h, n, 128, kb * 32, half);
    wf[8 + kb] = load_frag16(Whh1h, n, 128, kb * 32, half);
  }
  const float b1n = b1[n];

  // g0r = (x0 @ Wih0^T + b0) tile, register-resident: each lane computes and
  // keeps exactly the 8 C-elements it will re-add every step.
  float g0r[8];
  {
    v8f acc = {};
#pragma unroll
    for (int kb = 0; kb < 4; ++kb) {
      v16h a  = load_frag16(&afeat[0][0], l15, 128, kb * 32, half);
      v16h bq = load_frag16(Wih0h, n, 128, kb * 32, half);
      acc = WMMA_F16(a, bq, acc);
    }
    const float bn = b0[n];
#pragma unroll
    for (int r = 0; r < 8; ++r) g0r[r] = acc[r] + bn;
  }

  const float blv = bl[0];
  const int tstore = T - 50;

  for (int t = 0; t < T; ++t) {
    // layer-0 gates = g0r + h0 @ Whh0^T
    {
      v8f acc = {};
#pragma unroll
      for (int kb = 0; kb < 4; ++kb) {
        v16h a = load_frag16(&h0buf[0][0], l15, 128, kb * 32, half);
        acc = WMMA_F16(a, wf[kb], acc);
      }
#pragma unroll
      for (int r = 0; r < 8; ++r) gates[r + 8 * half][n] = acc[r] + g0r[r];
    }
    __syncthreads();
    for (int e = tid; e < 16 * 128; e += 1024) {  // layer-0 cell update
      const int bb = e >> 7, hh = e & 127;
      const float gi = gates[bb][hh], gf = gates[bb][128 + hh];
      const float gg = gates[bb][256 + hh], go = gates[bb][384 + hh];
      const float cn = sigmoidf_(gf) * c0s[bb][hh] + sigmoidf_(gi) * tanhf(gg);
      c0s[bb][hh] = cn;
      h0buf[bb][hh] = (_Float16)(sigmoidf_(go) * tanhf(cn));
    }
    __syncthreads();
    // layer-1 gates = h0' @ Wih1^T + h1 @ Whh1^T + b1
    {
      v8f acc = {};
#pragma unroll
      for (int kb = 0; kb < 4; ++kb) {
        v16h a = load_frag16(&h0buf[0][0], l15, 128, kb * 32, half);
        acc = WMMA_F16(a, wf[4 + kb], acc);
      }
#pragma unroll
      for (int kb = 0; kb < 4; ++kb) {
        v16h a = load_frag16(&h1buf[0][0], l15, 128, kb * 32, half);
        acc = WMMA_F16(a, wf[8 + kb], acc);
      }
#pragma unroll
      for (int r = 0; r < 8; ++r) gates[r + 8 * half][n] = acc[r] + b1n;
    }
    __syncthreads();
    for (int e = tid; e < 16 * 128; e += 1024) {  // layer-1 cell update
      const int bb = e >> 7, hh = e & 127;
      const float gi = gates[bb][hh], gf = gates[bb][128 + hh];
      const float gg = gates[bb][256 + hh], go = gates[bb][384 + hh];
      const float cn = sigmoidf_(gf) * c1s[bb][hh] + sigmoidf_(gi) * tanhf(gg);
      c1s[bb][hh] = cn;
      h1buf[bb][hh] = (_Float16)(sigmoidf_(go) * tanhf(cn));
    }
    __syncthreads();
    // predictions (last 50 seed the autoregressive window); reads h1buf only,
    // which is fenced above; next gates write is fenced by the same barrier.
    if (t >= tstore && tid < 16) {
      float s = blv;
#pragma unroll 16
      for (int hh = 0; hh < 128; ++hh) s += (float)h1buf[tid][hh] * wls[hh];
      winbuf[(size_t)(b0row + tid) * 50 + (t - tstore)] = s;
    }
  }
}

// ---------------------------------------------------------------------------
// K4: persistent autoregressive loop. grid=4 x 1024. Per step: collapsed-TCN
// feature via WMMA (K padded 50->64), LSTM step, scalar head, window shift.
// All weight fragments (Weff + 4 LSTM matrices) register-resident.
// ---------------------------------------------------------------------------
__global__ __launch_bounds__(1024) void phase_c(
    const float* __restrict__ winbuf,
    const _Float16* __restrict__ Weffh, const float* __restrict__ beff,
    const _Float16* __restrict__ Wih0h, const _Float16* __restrict__ Whh0h,
    const _Float16* __restrict__ Wih1h, const _Float16* __restrict__ Whh1h,
    const float* __restrict__ b0, const float* __restrict__ b1,
    const float* __restrict__ Wl, const float* __restrict__ bl,
    float* __restrict__ out, int NP) {
  __shared__ __align__(16) float    gates[16][512];
  __shared__ __align__(16) _Float16 awin[16][64];    // f16 window, K padded
  __shared__ float                  winf[16][50];    // f32 rolling window
  __shared__ __align__(16) _Float16 afeat[16][128];
  __shared__ __align__(16) _Float16 h0buf[16][128];
  __shared__ __align__(16) _Float16 h1buf[16][128];
  __shared__ float c0s[16][128], c1s[16][128];
  __shared__ float wls[128];

  const int tid   = threadIdx.x;
  const int lane  = tid & 31, l15 = lane & 15, half = lane >> 4;
  const int wave  = tid >> 5;
  const int b0row = blockIdx.x * 16;
  const int n     = wave * 16 + l15;

  for (int e = tid; e < 16 * 50; e += 1024)
    winf[e / 50][e % 50] = winbuf[(size_t)(b0row + e / 50) * 50 + (e % 50)];
  for (int e = tid; e < 16 * 128; e += 1024) {
    const int bb = e >> 7, hh = e & 127;
    h0buf[bb][hh] = (_Float16)0.f;
    h1buf[bb][hh] = (_Float16)0.f;
    c0s[bb][hh] = 0.f;
    c1s[bb][hh] = 0.f;
  }
  if (tid < 128) wls[tid] = Wl[tid];
  __syncthreads();

  // Loop-invariant weight fragments -> VGPRs.
  v16h wf[16];  // Wih0, Whh0, Wih1, Whh1 (4 k-blocks each)
#pragma unroll
  for (int kb = 0; kb < 4; ++kb) {
    wf[kb]      = load_frag16(Wih0h, n, 128, kb * 32, half);
    wf[4 + kb]  = load_frag16(Whh0h, n, 128, kb * 32, half);
    wf[8 + kb]  = load_frag16(Wih1h, n, 128, kb * 32, half);
    wf[12 + kb] = load_frag16(Whh1h, n, 128, kb * 32, half);
  }
  v16h wefff[2];  // feat GEMM B-fragments (row clamped; used by waves 0..7)
  const int nf = (wave & 7) * 16 + l15;
#pragma unroll
  for (int kb = 0; kb < 2; ++kb) wefff[kb] = load_frag16(Weffh, nf, 64, kb * 32, half);

  const float b0n = b0[n], b1n = b1[n];
  const float beffn = beff[nf];
  const float blv = bl[0];

  for (int s = 0; s < NP; ++s) {
    // pack window into f16 A-matrix (zero-pad K 50..63)
    for (int e = tid; e < 16 * 64; e += 1024) {
      const int bb = e >> 6, k = e & 63;
      awin[bb][k] = (k < 50) ? (_Float16)winf[bb][k] : (_Float16)0.f;
    }
    __syncthreads();
    // feat = relu(win @ Weff^T + beff): N=128 -> waves 0..7
    if (wave < 8) {
      v8f acc = {};
#pragma unroll
      for (int kb = 0; kb < 2; ++kb) {
        v16h a = load_frag16(&awin[0][0], l15, 64, kb * 32, half);
        acc = WMMA_F16(a, wefff[kb], acc);
      }
#pragma unroll
      for (int r = 0; r < 8; ++r) {
        const float x = acc[r] + beffn;
        afeat[r + 8 * half][nf] = (_Float16)(x > 0.f ? x : 0.f);
      }
    }
    __syncthreads();
    // layer-0 gates = feat @ Wih0^T + h0 @ Whh0^T + b0
    {
      v8f acc = {};
#pragma unroll
      for (int kb = 0; kb < 4; ++kb) {
        v16h a = load_frag16(&afeat[0][0], l15, 128, kb * 32, half);
        acc = WMMA_F16(a, wf[kb], acc);
      }
#pragma unroll
      for (int kb = 0; kb < 4; ++kb) {
        v16h a = load_frag16(&h0buf[0][0], l15, 128, kb * 32, half);
        acc = WMMA_F16(a, wf[4 + kb], acc);
      }
#pragma unroll
      for (int r = 0; r < 8; ++r) gates[r + 8 * half][n] = acc[r] + b0n;
    }
    __syncthreads();
    for (int e = tid; e < 16 * 128; e += 1024) {
      const int bb = e >> 7, hh = e & 127;
      const float gi = gates[bb][hh], gf = gates[bb][128 + hh];
      const float gg = gates[bb][256 + hh], go = gates[bb][384 + hh];
      const float cn = sigmoidf_(gf) * c0s[bb][hh] + sigmoidf_(gi) * tanhf(gg);
      c0s[bb][hh] = cn;
      h0buf[bb][hh] = (_Float16)(sigmoidf_(go) * tanhf(cn));
    }
    __syncthreads();
    // layer-1 gates = h0' @ Wih1^T + h1 @ Whh1^T + b1
    {
      v8f acc = {};
#pragma unroll
      for (int kb = 0; kb < 4; ++kb) {
        v16h a = load_frag16(&h0buf[0][0], l15, 128, kb * 32, half);
        acc = WMMA_F16(a, wf[8 + kb], acc);
      }
#pragma unroll
      for (int kb = 0; kb < 4; ++kb) {
        v16h a = load_frag16(&h1buf[0][0], l15, 128, kb * 32, half);
        acc = WMMA_F16(a, wf[12 + kb], acc);
      }
#pragma unroll
      for (int r = 0; r < 8; ++r) gates[r + 8 * half][n] = acc[r] + b1n;
    }
    __syncthreads();
    for (int e = tid; e < 16 * 128; e += 1024) {
      const int bb = e >> 7, hh = e & 127;
      const float gi = gates[bb][hh], gf = gates[bb][128 + hh];
      const float gg = gates[bb][256 + hh], go = gates[bb][384 + hh];
      const float cn = sigmoidf_(gf) * c1s[bb][hh] + sigmoidf_(gi) * tanhf(gg);
      c1s[bb][hh] = cn;
      h1buf[bb][hh] = (_Float16)(sigmoidf_(go) * tanhf(cn));
    }
    __syncthreads();
    // scalar head + window shift + emit
    if (tid < 16) {
      float v = blv;
#pragma unroll 16
      for (int hh = 0; hh < 128; ++hh) v += (float)h1buf[tid][hh] * wls[hh];
      for (int j = 0; j < 49; ++j) winf[tid][j] = winf[tid][j + 1];
      winf[tid][49] = v;
      out[(size_t)(b0row + tid) * NP + s] = v;
    }
    __syncthreads();  // winf stable before next window pack
  }
}

// ---------------------------------------------------------------------------
extern "C" void kernel_launch(void* const* d_in, const int* in_sizes, int n_in,
                              void* d_out, int out_size, void* d_ws, size_t ws_size,
                              hipStream_t stream) {
  const float* X    = (const float*)d_in[0];
  const float* w1   = (const float*)d_in[1];
  const float* cb1  = (const float*)d_in[2];
  const float* w2   = (const float*)d_in[3];
  const float* cb2  = (const float*)d_in[4];
  const float* w3   = (const float*)d_in[5];
  const float* cb3  = (const float*)d_in[6];
  const float* Wih0 = (const float*)d_in[7];
  const float* Whh0 = (const float*)d_in[8];
  const float* bih0 = (const float*)d_in[9];
  const float* bhh0 = (const float*)d_in[10];
  const float* Wih1 = (const float*)d_in[11];
  const float* Whh1 = (const float*)d_in[12];
  const float* bih1 = (const float*)d_in[13];
  const float* bhh1 = (const float*)d_in[14];
  const float* Wl   = (const float*)d_in[15];
  const float* bl   = (const float*)d_in[16];
  (void)n_in; (void)ws_size;

  const int B  = 64;
  const int L  = in_sizes[0] / B;  // 2048
  const int T  = L - 49;           // conv FOV-trimmed length (1999)
  const int NP = out_size / B;     // num_predictions (128)

  char* ws = (char*)d_ws;
  size_t off = 0;
  auto take = [&](size_t bytes) -> char* {
    off = (off + 255) & ~(size_t)255;
    char* p = ws + off;
    off += bytes;
    return p;
  };
  _Float16* Wih0h  = (_Float16*)take(512 * 128 * sizeof(_Float16));
  _Float16* Whh0h  = (_Float16*)take(512 * 128 * sizeof(_Float16));
  _Float16* Wih1h  = (_Float16*)take(512 * 128 * sizeof(_Float16));
  _Float16* Whh1h  = (_Float16*)take(512 * 128 * sizeof(_Float16));
  _Float16* Weffh  = (_Float16*)take(128 * 64 * sizeof(_Float16));
  float*    Weff32 = (float*)take(128 * 50 * sizeof(float));
  float*    beff   = (float*)take(128 * sizeof(float));
  float*    b0f    = (float*)take(512 * sizeof(float));
  float*    b1f    = (float*)take(512 * sizeof(float));
  float*    winbuf = (float*)take((size_t)B * 50 * sizeof(float));

  compose_weights<<<dim3(128), dim3(256), 0, stream>>>(
      w1, cb1, w2, cb2, w3, cb3, Weff32, Weffh, beff);
  convert_lstm<<<dim3(256), dim3(256), 0, stream>>>(
      Wih0, Whh0, Wih1, Whh1, bih0, bhh0, bih1, bhh1,
      Wih0h, Whh0h, Wih1h, Whh1h, b0f, b1f);
  phase_b<<<dim3(4), dim3(1024), 0, stream>>>(
      X, L, T, Weff32, beff, Wih0h, Whh0h, Wih1h, Whh1h,
      b0f, b1f, Wl, bl, winbuf);
  phase_c<<<dim3(4), dim3(1024), 0, stream>>>(
      winbuf, Weffh, beff, Wih0h, Whh0h, Wih1h, Whh1h,
      b0f, b1f, Wl, bl, (float*)d_out, NP);
}
